// MoELayer_21294447853711
// MI455X (gfx1250) — matrix operand order repaired and checked by reading, not compiled, not run on hardware
//
#include <hip/hip_runtime.h>

#define NTOK   16384
#define HID    1024
#define DFF_   4096
#define NEXP   8
#define DC     1024          // DFF chunk processed per GEMM1/GEMM2 pass
#define NCHUNK (DFF_ / DC)
#define KT     64            // K-extent of one LDS stage (2 WMMA k-steps)
#define LDB    72            // LDS row stride in halfs (144B: 16B-aligned, bank-spread)
#define USE_ASYNC_LDS 1

typedef __bf16 v16bf __attribute__((ext_vector_type(16)));
typedef float  v8f   __attribute__((ext_vector_type(8)));

union Frag { v16bf v; uint4 q[2]; };

__device__ __forceinline__ unsigned short f2bf(float f) {
  unsigned u = __builtin_bit_cast(unsigned, f);
  u += 0x7FFFu + ((u >> 16) & 1u);   // round-to-nearest-even
  return (unsigned short)(u >> 16);
}
// pack two floats -> two bf16 in one dword (lo half = first element)
__device__ __forceinline__ unsigned pack2bf(float lo, float hi) {
  return ((unsigned)f2bf(hi) << 16) | (unsigned)f2bf(lo);
}

// A fragment: 16x32 bf16 from row-major LDS tile at k-offset ks.
// Lane l: M = l&15; halves 0..7 = K(ks + (l<16?0:8)+i), halves 8..15 = K(ks + (l<16?16:24)+i).
__device__ __forceinline__ v16bf load_a_frag(const unsigned short* As, int rowBase, int ks, int lane) {
  const unsigned short* row = As + (size_t)(rowBase + (lane & 15)) * LDB + ks;
  int k0 = (lane < 16) ? 0 : 8;
  int k1 = (lane < 16) ? 16 : 24;
  Frag u;
  u.q[0] = *(const uint4*)(row + k0);
  u.q[1] = *(const uint4*)(row + k1);
  return u.v;
}

// B fragment: 32x16 bf16 from K-contiguous (transposed) LDS tile Bs[n][k].
// Lane l: N = l&15; halves 0..7 = K(ks + (l<16?0:8)+i), halves 8..15 = K(ks + (l<16?16:24)+i).
__device__ __forceinline__ v16bf load_b_frag(const unsigned short* Bs, int nBase, int ks, int lane) {
  const unsigned short* col = Bs + (size_t)(nBase + (lane & 15)) * LDB + ks + ((lane < 16) ? 0 : 8);
  Frag u;
  u.q[0] = *(const uint4*)(col);
  u.q[1] = *(const uint4*)(col + 16);
  return u.v;
}

__global__ void init_counts(int* counts) {
  if (threadIdx.x < NEXP) counts[threadIdx.x] = 0;
}

// One wave per token: gate logits, softmax weight of argmax, compaction lists.
__global__ __launch_bounds__(256) void gate_kernel(const float* __restrict__ x,
                                                   const float* __restrict__ gw,
                                                   float* __restrict__ ew,
                                                   int* __restrict__ counts,
                                                   int* __restrict__ list) {
  int lane = threadIdx.x & 31;
  int t = blockIdx.x * 8 + (threadIdx.x >> 5);
  if (t >= NTOK) return;

  float acc[NEXP];
#pragma unroll
  for (int e = 0; e < NEXP; ++e) acc[e] = 0.f;

  const float* xr = x + (size_t)t * HID;
#pragma unroll 4
  for (int i = 0; i < HID / 32; ++i) {
    int h = lane + i * 32;
    float xv = xr[h];
#pragma unroll
    for (int e = 0; e < NEXP; ++e) acc[e] += xv * gw[e * HID + h];
  }
#pragma unroll
  for (int e = 0; e < NEXP; ++e) {
#pragma unroll
    for (int off = 16; off > 0; off >>= 1) acc[e] += __shfl_xor(acc[e], off, 32);
  }
  if (lane == 0) {
    float mx = acc[0]; int mi = 0;
#pragma unroll
    for (int e = 1; e < NEXP; ++e) { if (acc[e] > mx) { mx = acc[e]; mi = e; } }
    float s = 0.f;
#pragma unroll
    for (int e = 0; e < NEXP; ++e) s += __expf(acc[e] - mx);
    ew[t] = 1.0f / s;                       // prob of argmax expert
    int pos = atomicAdd(&counts[mi], 1);
    list[mi * NTOK + pos] = t;
  }
}

__global__ void scan_counts(const int* __restrict__ counts, int* __restrict__ offs) {
  if (threadIdx.x == 0) {
    int s = 0;
    for (int e = 0; e < NEXP; ++e) { offs[e] = s; s += counts[e]; }
  }
}

// GEMM1: h1[rows(e), DC cols] = relu( X[gathered] @ W1[e][:, chunk] + b1 )  (bf16 out)
__global__ __launch_bounds__(256) void ffn1_kernel(const float* __restrict__ x,
                                                   const float* __restrict__ w1,
                                                   const float* __restrict__ b1,
                                                   const int* __restrict__ counts,
                                                   const int* __restrict__ offs,
                                                   const int* __restrict__ list,
                                                   unsigned short* __restrict__ h1,
                                                   int cchunk) {
  int e = blockIdx.z;
  int cnt = counts[e];
  int mbase = blockIdx.x * 128;
  if (mbase >= cnt) return;
  int nloc = blockIdx.y * 128;        // column within chunk [0, DC)
  int ncol0 = cchunk * DC + nloc;     // global DFF column

  __shared__ unsigned short As[128 * LDB];
  __shared__ unsigned short Bs[128 * LDB];
  __shared__ int toks[128];

  int tid = threadIdx.x;
  if (tid < 128) toks[tid] = list[e * NTOK + min(mbase + tid, cnt - 1)];
  __syncthreads();

  int lane = tid & 31, wv = tid >> 5;
  int wmBase = (wv & 1) * 64;
  int wnBase = (wv >> 1) * 32;
  const float* w1e = w1 + (size_t)e * HID * DFF_;

  v8f acc[4][2] = {};

  for (int k0 = 0; k0 < HID; k0 += KT) {
    { // A tile: gathered token rows, fp32 -> packed bf16, 2x ds_store_b128 x2
      int r  = tid >> 1;
      int kk = (tid & 1) << 5;                 // 32 halfs per thread
      const float* src = x + (size_t)toks[r] * HID + k0 + kk;
      unsigned q[16];
#pragma unroll
      for (int j = 0; j < 8; ++j) {
        float4 p = *(const float4*)(src + 4 * j);
        q[2 * j + 0] = pack2bf(p.x, p.y);
        q[2 * j + 1] = pack2bf(p.z, p.w);
      }
      uint4* dst = (uint4*)&As[r * LDB + kk];
#pragma unroll
      for (int j = 0; j < 4; ++j)
        dst[j] = make_uint4(q[4 * j], q[4 * j + 1], q[4 * j + 2], q[4 * j + 3]);
    }
    { // B tile: W1 rows -> transposed LDS Bs[f][k], packed, 4x ds_store_b128
      int f0 = (tid & 31) << 2;                // 4 f columns
      int kb = (tid >> 5) << 3;                // 8 k rows
      const float* src = w1e + (size_t)(k0 + kb) * DFF_ + ncol0 + f0;
      unsigned q[4][4];
#pragma unroll
      for (int kk2 = 0; kk2 < 4; ++kk2) {
        float4 p0 = *(const float4*)(src + (size_t)(2 * kk2) * DFF_);
        float4 p1 = *(const float4*)(src + (size_t)(2 * kk2 + 1) * DFF_);
        q[0][kk2] = pack2bf(p0.x, p1.x);
        q[1][kk2] = pack2bf(p0.y, p1.y);
        q[2][kk2] = pack2bf(p0.z, p1.z);
        q[3][kk2] = pack2bf(p0.w, p1.w);
      }
#pragma unroll
      for (int f = 0; f < 4; ++f)
        *(uint4*)&Bs[(f0 + f) * LDB + kb] =
            make_uint4(q[f][0], q[f][1], q[f][2], q[f][3]);
    }
    __syncthreads();
#pragma unroll
    for (int ks = 0; ks < KT; ks += 32) {
      v16bf af[4], bfr[2];
#pragma unroll
      for (int mt = 0; mt < 4; ++mt) af[mt] = load_a_frag(As, wmBase + mt * 16, ks, lane);
#pragma unroll
      for (int nt = 0; nt < 2; ++nt) bfr[nt] = load_b_frag(Bs, wnBase + nt * 16, ks, lane);
#pragma unroll
      for (int mt = 0; mt < 4; ++mt) {
#pragma unroll
        for (int nt = 0; nt < 2; ++nt) {
          acc[mt][nt] = __builtin_amdgcn_wmma_f32_16x16x32_bf16(
              false, af[mt], false, bfr[nt], (short)0, acc[mt][nt], false, false);
        }
      }
    }
    __syncthreads();
  }

  int base = offs[e];
#pragma unroll
  for (int mt = 0; mt < 4; ++mt) {
#pragma unroll
    for (int nt = 0; nt < 2; ++nt) {
      int n = wnBase + nt * 16 + (lane & 15);
      float bias = b1[e * DFF_ + ncol0 + n];
      int rb = wmBase + mt * 16 + ((lane >> 4) << 3);
#pragma unroll
      for (int rr = 0; rr < 8; ++rr) {
        int rowid = mbase + rb + rr;
        if (rowid < cnt) {
          float v = acc[mt][nt][rr] + bias;
          v = v > 0.f ? v : 0.f;
          h1[(size_t)(base + rowid) * DC + nloc + n] = f2bf(v);
        }
      }
    }
  }
}

// GEMM2: out[token] (chunk0: = w*(acc+b2), else += w*acc) from h1 @ W2[e][chunk,:]
__global__ __launch_bounds__(256) void ffn2_kernel(const float* __restrict__ w2,
                                                   const float* __restrict__ b2,
                                                   const unsigned short* __restrict__ h1,
                                                   const float* __restrict__ ew,
                                                   const int* __restrict__ counts,
                                                   const int* __restrict__ offs,
                                                   const int* __restrict__ list,
                                                   float* __restrict__ out,
                                                   int cchunk) {
  int e = blockIdx.z;
  int cnt = counts[e];
  int mbase = blockIdx.x * 128;
  if (mbase >= cnt) return;
  int nbase = blockIdx.y * 128;       // H column base

  __shared__ unsigned short As[128 * LDB];
  __shared__ unsigned short Bs[128 * LDB];
  __shared__ int toks[128];

  int tid = threadIdx.x;
  if (tid < 128) toks[tid] = list[e * NTOK + min(mbase + tid, cnt - 1)];
  __syncthreads();

  int lane = tid & 31, wv = tid >> 5;
  int wmBase = (wv & 1) * 64;
  int wnBase = (wv >> 1) * 32;
  int base = offs[e];
  const float* w2e = w2 + (size_t)e * DFF_ * HID;

  v8f acc[4][2] = {};

  for (int k0 = 0; k0 < DC; k0 += KT) {
    { // A tile: bf16 activations, global -> LDS directly (ASYNCcnt path)
      int r  = tid >> 1;
      int kk = (tid & 1) << 5;                 // 32 halfs = 64B per thread
      const unsigned short* src =
          h1 + (size_t)(base + min(mbase + r, cnt - 1)) * DC + k0 + kk;
#if USE_ASYNC_LDS
      unsigned           lds0 = (unsigned)(size_t)&As[r * LDB + kk];
      unsigned long long g0   = (unsigned long long)(size_t)src;
#pragma unroll
      for (int j = 0; j < 4; ++j) {
        asm volatile("global_load_async_to_lds_b128 %0, %1, off"
                     :: "v"(lds0 + 16u * j), "v"(g0 + 16ull * j) : "memory");
      }
#else
      uint4* dst = (uint4*)&As[r * LDB + kk];
#pragma unroll
      for (int j = 0; j < 4; ++j) dst[j] = ((const uint4*)src)[j];
#endif
    }
    { // B tile: W2 rows -> transposed LDS, packed, 4x ds_store_b128
      int f0 = (tid & 31) << 2;
      int kb = (tid >> 5) << 3;
      const float* src = w2e + (size_t)(cchunk * DC + k0 + kb) * HID + nbase + f0;
      unsigned q[4][4];
#pragma unroll
      for (int kk2 = 0; kk2 < 4; ++kk2) {
        float4 p0 = *(const float4*)(src + (size_t)(2 * kk2) * HID);
        float4 p1 = *(const float4*)(src + (size_t)(2 * kk2 + 1) * HID);
        q[0][kk2] = pack2bf(p0.x, p1.x);
        q[1][kk2] = pack2bf(p0.y, p1.y);
        q[2][kk2] = pack2bf(p0.z, p1.z);
        q[3][kk2] = pack2bf(p0.w, p1.w);
      }
#pragma unroll
      for (int f = 0; f < 4; ++f)
        *(uint4*)&Bs[(f0 + f) * LDB + kb] =
            make_uint4(q[f][0], q[f][1], q[f][2], q[f][3]);
    }
#if USE_ASYNC_LDS
    asm volatile("s_wait_asynccnt 0x0" ::: "memory");
#endif
    __syncthreads();
#pragma unroll
    for (int ks = 0; ks < KT; ks += 32) {
      v16bf af[4], bfr[2];
#pragma unroll
      for (int mt = 0; mt < 4; ++mt) af[mt] = load_a_frag(As, wmBase + mt * 16, ks, lane);
#pragma unroll
      for (int nt = 0; nt < 2; ++nt) bfr[nt] = load_b_frag(Bs, wnBase + nt * 16, ks, lane);
#pragma unroll
      for (int mt = 0; mt < 4; ++mt) {
#pragma unroll
        for (int nt = 0; nt < 2; ++nt) {
          acc[mt][nt] = __builtin_amdgcn_wmma_f32_16x16x32_bf16(
              false, af[mt], false, bfr[nt], (short)0, acc[mt][nt], false, false);
        }
      }
    }
    __syncthreads();
  }

#pragma unroll
  for (int mt = 0; mt < 4; ++mt) {
#pragma unroll
    for (int nt = 0; nt < 2; ++nt) {
      int n = wnBase + nt * 16 + (lane & 15);
      int ncol = nbase + n;
      float b2v = b2[e * HID + ncol];
      int rb = wmBase + mt * 16 + ((lane >> 4) << 3);
#pragma unroll
      for (int rr = 0; rr < 8; ++rr) {
        int rowid = mbase + rb + rr;
        if (rowid < cnt) {
          int t = toks[rb + rr];
          float w = ew[t];
          float v = acc[mt][nt][rr];
          float* o = out + (size_t)t * HID + ncol;
          if (cchunk == 0) *o = w * (v + b2v);
          else             *o += w * v;
        }
      }
    }
  }
}

extern "C" void kernel_launch(void* const* d_in, const int* in_sizes, int n_in,
                              void* d_out, int out_size, void* d_ws, size_t ws_size,
                              hipStream_t stream) {
  const float* x  = (const float*)d_in[0];   // hidden_states [B,S,H] fp32
  const float* gw = (const float*)d_in[1];   // gate_w [E,H]
  const float* w1 = (const float*)d_in[2];   // [E,H,DFF]
  const float* b1 = (const float*)d_in[3];   // [E,DFF]
  const float* w2 = (const float*)d_in[4];   // [E,DFF,H]
  const float* b2 = (const float*)d_in[5];   // [E,H]
  float* out = (float*)d_out;

  char* ws = (char*)d_ws;
  int*            counts = (int*)(ws + 0);          //  8 ints
  int*            offs   = (int*)(ws + 64);         //  8 ints
  float*          ew     = (float*)(ws + 1024);     //  N floats  (64 KB)
  int*            list   = (int*)(ws + (1 << 17));  //  E*N ints  (512 KB)
  unsigned short* h1     = (unsigned short*)(ws + (1 << 20)); // N*DC bf16 (33.5 MB)

  hipLaunchKernelGGL(init_counts, dim3(1), dim3(32), 0, stream, counts);
  hipLaunchKernelGGL(gate_kernel, dim3(NTOK / 8), dim3(256), 0, stream,
                     x, gw, ew, counts, list);
  hipLaunchKernelGGL(scan_counts, dim3(1), dim3(32), 0, stream, counts, offs);

  for (int c = 0; c < NCHUNK; ++c) {
    hipLaunchKernelGGL(ffn1_kernel, dim3(NTOK / 128, DC / 128, NEXP), dim3(256), 0, stream,
                       x, w1, b1, counts, offs, list, h1, c);
    hipLaunchKernelGGL(ffn2_kernel, dim3(NTOK / 128, HID / 128, NEXP), dim3(256), 0, stream,
                       w2, b2, h1, ew, counts, offs, list, out, c);
  }

  (void)in_sizes; (void)n_in; (void)out_size; (void)ws_size;
}
